// MultiHeadAttention_22170621182116
// MI455X (gfx1250) — matrix operand order repaired and checked by reading, not compiled, not run on hardware
//
#include <hip/hip_runtime.h>
#include <hip/hip_bf16.h>

#define EMB 1024
#define NH  16
#define HD  64
#define TT  2048
#define BB  2

typedef __attribute__((ext_vector_type(16))) _Float16 v16h;
typedef __attribute__((ext_vector_type(8)))  _Float16 v8h;
typedef __attribute__((ext_vector_type(2)))  _Float16 h2;
typedef __attribute__((ext_vector_type(8)))  float    v8f;
typedef __attribute__((ext_vector_type(4)))  float    v4f;

__device__ __forceinline__ v16h cat8(v8h lo, v8h hi) {
    v16h r;
#pragma unroll
    for (int i = 0; i < 8; ++i) { r[i] = lo[i]; r[i + 8] = hi[i]; }
    return r;
}

__device__ __forceinline__ v8f wmma_f16(v16h a, v16h b, v8f c) {
    return __builtin_amdgcn_wmma_f32_16x16x32_f16(false, a, false, b, (short)0, c, false, false);
}

// packed f32x2 -> f16x2 (v_cvt_pk_f16_f32 rtz flavor)
__device__ __forceinline__ h2 pk(float a, float b) {
    auto r = __builtin_amdgcn_cvt_pkrtz(a, b);
    return *(h2*)&r;
}

// async copy of 32 contiguous bytes (f16 data) from global to LDS
__device__ __forceinline__ void async_copy32(_Float16* lds_dst, const _Float16* gsrc) {
    unsigned la = (unsigned)(unsigned long long)lds_dst;   // LDS byte offset
    asm volatile("global_load_async_to_lds_b128 %0, %1, off"
                 :: "v"(la), "v"(gsrc) : "memory");
    asm volatile("global_load_async_to_lds_b128 %0, %1, off offset:16"
                 :: "v"(la), "v"(gsrc) : "memory");
}

__device__ __forceinline__ void wait_async0() {
    asm volatile("s_wait_asynccnt 0" ::: "memory");
}

// ---------------------------------------------------------------------------
// Kernel 1: qkv = x @ Wqkv  (fp32 in, f16 out, head-major [B,H,T,64] layout)
// ---------------------------------------------------------------------------
__global__ __launch_bounds__(128)
void qkv_gemm_kernel(const float* __restrict__ x, const float* __restrict__ Wqkv,
                     _Float16* __restrict__ qf, _Float16* __restrict__ kf,
                     _Float16* __restrict__ vf)
{
    __shared__ alignas(64) _Float16 As[64][32];   // [m][k]
    __shared__ alignas(64) _Float16 Bs[64][32];   // [n][k] (transposed)
    const int tid = threadIdx.x;
    const int w = tid >> 5, lane = tid & 31, l16 = lane & 15, hi = (lane >> 4) & 1;
    const int bm0 = blockIdx.x * 64;
    const int bn0 = blockIdx.y * 64;
    v8f acc[4] = {};

    for (int kk = 0; kk < EMB / 32; ++kk) {
        __syncthreads();
        {   // A tile: 64x32 fp32 -> f16 (packed cvt, vector LDS stores)
            int row = tid >> 1;
            int c0  = (tid & 1) * 16;
            const float* src = x + (size_t)(bm0 + row) * EMB + kk * 32 + c0;
            float va[16];
#pragma unroll
            for (int j = 0; j < 16; j += 4) {
                v4f v = *(const v4f*)(src + j);
                va[j] = v[0]; va[j + 1] = v[1]; va[j + 2] = v[2]; va[j + 3] = v[3];
            }
            v8h o0, o1;
#pragma unroll
            for (int j = 0; j < 8; j += 2) {
                h2 p = pk(va[j], va[j + 1]);     o0[j] = p[0]; o0[j + 1] = p[1];
                h2 q = pk(va[8 + j], va[9 + j]); o1[j] = q[0]; o1[j + 1] = q[1];
            }
            *(v8h*)&As[row][c0]     = o0;
            *(v8h*)&As[row][c0 + 8] = o1;
        }
        {   // B tile: 32x64 fp32, transposed store [n][k] with paired k (b32 stores)
            int k2 = (tid & 15) * 2;          // k pair
            int n0 = (tid >> 4) * 8;          // 8 columns
            const float* s0 = Wqkv + (size_t)(kk * 32 + k2) * (3 * EMB) + bn0 + n0;
            const float* s1 = s0 + 3 * EMB;
#pragma unroll
            for (int j = 0; j < 8; j += 4) {
                v4f a = *(const v4f*)(s0 + j);
                v4f b = *(const v4f*)(s1 + j);
                *(h2*)&Bs[n0 + j + 0][k2] = pk(a[0], b[0]);
                *(h2*)&Bs[n0 + j + 1][k2] = pk(a[1], b[1]);
                *(h2*)&Bs[n0 + j + 2][k2] = pk(a[2], b[2]);
                *(h2*)&Bs[n0 + j + 3][k2] = pk(a[3], b[3]);
            }
        }
        __syncthreads();
        const int arow = w * 16 + l16;
        v16h af = cat8(*(const v8h*)&As[arow][hi * 8],
                       *(const v8h*)&As[arow][16 + hi * 8]);
#pragma unroll
        for (int nt = 0; nt < 4; ++nt) {
            v16h bf = cat8(*(const v8h*)&Bs[nt * 16 + l16][hi * 8], *(const v8h*)&Bs[nt * 16 + l16][16 + hi * 8]);
            acc[nt] = wmma_f16(af, bf, acc[nt]);
        }
    }

#pragma unroll
    for (int nt = 0; nt < 4; ++nt) {
#pragma unroll
        for (int r = 0; r < 8; ++r) {
            int row = bm0 + w * 16 + r + hi * 8;
            int col = bn0 + nt * 16 + l16;
            int b = row >> 11, t = row & (TT - 1);
            int which = col >> 10, rem = col & (EMB - 1);
            int h = rem >> 6, d = rem & (HD - 1);
            size_t idx = ((size_t)((b * NH + h) * TT + t)) * HD + d;
            _Float16 v = (_Float16)acc[nt][r];
            if (which == 0)      qf[idx] = v;
            else if (which == 1) kf[idx] = v;
            else                 vf[idx] = v;
        }
    }
}

// ---------------------------------------------------------------------------
// Kernel 2: RoPE on q,k (in-place, f16); folds softmax scale 1/sqrt(64) into q
// ---------------------------------------------------------------------------
__global__ __launch_bounds__(256)
void rope_kernel(_Float16* __restrict__ qf, _Float16* __restrict__ kf,
                 const float* __restrict__ cosb, const float* __restrict__ sinb)
{
    int idx = blockIdx.x * blockDim.x + threadIdx.x;
    int d  = idx & 31;
    int t  = (idx >> 5) & (TT - 1);
    int bh = idx >> 16;
    float c = cosb[t * 32 + d], s = sinb[t * 32 + d];
    size_t base = ((size_t)bh * TT + t) * HD + 2 * d;
    float qe = (float)qf[base], qo = (float)qf[base + 1];
    float ke = (float)kf[base], ko = (float)kf[base + 1];
    const float sc = 0.125f;
    *(h2*)&qf[base] = pk((qe * c - qo * s) * sc, (qe * s + qo * c) * sc);
    *(h2*)&kf[base] = pk(ke * c - ko * s, ke * s + ko * c);
}

// ---------------------------------------------------------------------------
// Kernel 3: causal flash attention, one (b,h) x 64-query tile per workgroup
// ---------------------------------------------------------------------------
__global__ __launch_bounds__(128)
void flash_attn_kernel(const _Float16* __restrict__ qf, const _Float16* __restrict__ kf,
                       const _Float16* __restrict__ vf, _Float16* __restrict__ attn)
{
    __shared__ alignas(64) _Float16 Qs[64][64];       // [q][dim]
    __shared__ alignas(64) _Float16 Kt[64][64];       // [key][dim]
    __shared__ alignas(64) _Float16 Vt[64][64];       // [dim][key]
    __shared__ alignas(64) _Float16 Ps[4][16][64];    // per-wave P staging
    const int tid = threadIdx.x;
    const int w = tid >> 5, lane = tid & 31, l16 = lane & 15, hi = (lane >> 4) & 1;
    const int qbase = blockIdx.x * 64;
    const int bh = blockIdx.y;
    const _Float16* qp = qf + (size_t)bh * TT * HD;
    const _Float16* kp = kf + (size_t)bh * TT * HD;
    const _Float16* vp = vf + (size_t)bh * TT * HD;

    {   // async load Q tile into LDS (f16, no conversion needed)
        int row = tid >> 1, half = tid & 1;
        async_copy32(&Qs[row][half * 32], qp + (size_t)(qbase + row) * HD + half * 32);
    }
    wait_async0();
    __syncthreads();
    const int qrow = w * 16 + l16;
    v16h qa0 = cat8(*(const v8h*)&Qs[qrow][hi * 8],      *(const v8h*)&Qs[qrow][16 + hi * 8]);
    v16h qa1 = cat8(*(const v8h*)&Qs[qrow][32 + hi * 8], *(const v8h*)&Qs[qrow][48 + hi * 8]);

    float m_[8], l_[8];
    v8f o[4] = {};
#pragma unroll
    for (int r = 0; r < 8; ++r) { m_[r] = -__builtin_inff(); l_[r] = 0.f; }

    const int ktiles = qbase / 64 + 1;
    for (int kt = 0; kt < ktiles; ++kt) {
        const int kbase = kt * 64;
        __syncthreads();
        {   // K tile: async straight to LDS [key][dim]
            int key = tid >> 1, half = tid & 1;
            async_copy32(&Kt[key][half * 32], kp + (size_t)(kbase + key) * HD + half * 32);
        }
        {   // V tile: load pair of key rows, transpose-store as packed pairs [dim][key]
            int kp2  = (tid & 31) * 2;        // key pair
            int dseg = (tid >> 5) * 16;       // 16 dims
            const _Float16* s0 = vp + (size_t)(kbase + kp2) * HD + dseg;
            v16h a = *(const v16h*)s0;
            v16h b = *(const v16h*)(s0 + HD);
#pragma unroll
            for (int j = 0; j < 16; ++j) {
                h2 p; p[0] = a[j]; p[1] = b[j];
                *(h2*)&Vt[dseg + j][kp2] = p;
            }
        }
        wait_async0();
        __syncthreads();

        // S = Q * K^T
        v8f s[4] = {};
#pragma unroll
        for (int nt = 0; nt < 4; ++nt) {
            v16h b0 = cat8(*(const v8h*)&Kt[nt * 16 + l16][hi * 8], *(const v8h*)&Kt[nt * 16 + l16][16 + hi * 8]);
            v16h b1 = cat8(*(const v8h*)&Kt[nt * 16 + l16][32 + hi * 8], *(const v8h*)&Kt[nt * 16 + l16][48 + hi * 8]);
            s[nt] = wmma_f16(qa0, b0, s[nt]);
            s[nt] = wmma_f16(qa1, b1, s[nt]);
        }

        if (kt == ktiles - 1) {  // causal mask on diagonal tile
#pragma unroll
            for (int nt = 0; nt < 4; ++nt)
#pragma unroll
                for (int r = 0; r < 8; ++r) {
                    int q = qbase + w * 16 + r + hi * 8;
                    int k = kbase + nt * 16 + l16;
                    if (k > q) s[nt][r] = -__builtin_inff();
                }
        }

        // online softmax (rows live inside one 16-lane half)
#pragma unroll
        for (int r = 0; r < 8; ++r) {
            float mx = fmaxf(fmaxf(s[0][r], s[1][r]), fmaxf(s[2][r], s[3][r]));
            mx = fmaxf(mx, __shfl_xor(mx, 1));
            mx = fmaxf(mx, __shfl_xor(mx, 2));
            mx = fmaxf(mx, __shfl_xor(mx, 4));
            mx = fmaxf(mx, __shfl_xor(mx, 8));
            float mnew = fmaxf(m_[r], mx);
            float scal = __expf(m_[r] - mnew);
            float rs = 0.f;
#pragma unroll
            for (int nt = 0; nt < 4; ++nt) {
                float p = __expf(s[nt][r] - mnew);
                s[nt][r] = p;
                rs += p;
            }
            rs += __shfl_xor(rs, 1); rs += __shfl_xor(rs, 2);
            rs += __shfl_xor(rs, 4); rs += __shfl_xor(rs, 8);
            l_[r] = l_[r] * scal + rs;
            m_[r] = mnew;
#pragma unroll
            for (int nt = 0; nt < 4; ++nt) o[nt][r] *= scal;
        }

        // stage P through per-wave LDS (C-layout -> A-layout), scalar f16 stores
#pragma unroll
        for (int nt = 0; nt < 4; ++nt)
#pragma unroll
            for (int r = 0; r < 8; r += 2) {
                h2 p = pk(s[nt][r], 0.f);   // row r
                Ps[w][r + 0 + hi * 8][nt * 16 + l16] = p[0];
                Ps[w][r + 1 + hi * 8][nt * 16 + l16] = (_Float16)s[nt][r + 1];
            }
        asm volatile("s_wait_dscnt 0" ::: "memory");
        v16h pa0 = cat8(*(const v8h*)&Ps[w][l16][hi * 8],      *(const v8h*)&Ps[w][l16][16 + hi * 8]);
        v16h pa1 = cat8(*(const v8h*)&Ps[w][l16][32 + hi * 8], *(const v8h*)&Ps[w][l16][48 + hi * 8]);

        // O += P * V
#pragma unroll
        for (int nt = 0; nt < 4; ++nt) {
            v16h vb0 = cat8(*(const v8h*)&Vt[nt * 16 + l16][hi * 8], *(const v8h*)&Vt[nt * 16 + l16][16 + hi * 8]);
            v16h vb1 = cat8(*(const v8h*)&Vt[nt * 16 + l16][32 + hi * 8], *(const v8h*)&Vt[nt * 16 + l16][48 + hi * 8]);
            o[nt] = wmma_f16(pa0, vb0, o[nt]);
            o[nt] = wmma_f16(pa1, vb1, o[nt]);
        }
    }

    // finalize: divide by l, write attn rows [B*T][EMB] (cols h*64+d)
#pragma unroll
    for (int r = 0; r < 8; ++r) {
        float inv = 1.0f / l_[r];
        int q = qbase + w * 16 + r + hi * 8;
        int b = bh >> 4, h = bh & (NH - 1);
        size_t rowbase = ((size_t)(b * TT + q)) * EMB + h * HD;
#pragma unroll
        for (int nt = 0; nt < 4; ++nt)
            attn[rowbase + nt * 16 + l16] = (_Float16)(o[nt][r] * inv);
    }
}

// ---------------------------------------------------------------------------
// Kernel 4: out = attn @ Wproj + bproj  (A async-loaded f16, fp32 out)
// ---------------------------------------------------------------------------
__global__ __launch_bounds__(128)
void proj_gemm_kernel(const _Float16* __restrict__ attn, const float* __restrict__ Wproj,
                      const float* __restrict__ bproj, float* __restrict__ out)
{
    __shared__ alignas(64) _Float16 As[64][32];
    __shared__ alignas(64) _Float16 Bs[64][32];   // [n][k]
    const int tid = threadIdx.x;
    const int w = tid >> 5, lane = tid & 31, l16 = lane & 15, hi = (lane >> 4) & 1;
    const int bm0 = blockIdx.x * 64;
    const int bn0 = blockIdx.y * 64;
    v8f acc[4] = {};

    for (int kk = 0; kk < EMB / 32; ++kk) {
        __syncthreads();
        {   // A tile: already f16 -> async copy to LDS
            int row = tid >> 1, c0 = (tid & 1) * 16;
            async_copy32(&As[row][c0], attn + (size_t)(bm0 + row) * EMB + kk * 32 + c0);
        }
        {   // B tile: transposed store with paired k, packed cvt
            int k2 = (tid & 15) * 2;
            int n0 = (tid >> 4) * 8;
            const float* s0 = Wproj + (size_t)(kk * 32 + k2) * EMB + bn0 + n0;
            const float* s1 = s0 + EMB;
#pragma unroll
            for (int j = 0; j < 8; j += 4) {
                v4f a = *(const v4f*)(s0 + j);
                v4f b = *(const v4f*)(s1 + j);
                *(h2*)&Bs[n0 + j + 0][k2] = pk(a[0], b[0]);
                *(h2*)&Bs[n0 + j + 1][k2] = pk(a[1], b[1]);
                *(h2*)&Bs[n0 + j + 2][k2] = pk(a[2], b[2]);
                *(h2*)&Bs[n0 + j + 3][k2] = pk(a[3], b[3]);
            }
        }
        wait_async0();
        __syncthreads();
        const int arow = w * 16 + l16;
        v16h af = cat8(*(const v8h*)&As[arow][hi * 8],
                       *(const v8h*)&As[arow][16 + hi * 8]);
#pragma unroll
        for (int nt = 0; nt < 4; ++nt) {
            v16h bf = cat8(*(const v8h*)&Bs[nt * 16 + l16][hi * 8], *(const v8h*)&Bs[nt * 16 + l16][16 + hi * 8]);
            acc[nt] = wmma_f16(af, bf, acc[nt]);
        }
    }

#pragma unroll
    for (int nt = 0; nt < 4; ++nt)
#pragma unroll
        for (int r = 0; r < 8; ++r) {
            int row = bm0 + w * 16 + r + hi * 8;
            int col = bn0 + nt * 16 + l16;
            out[(size_t)row * EMB + col] = acc[nt][r] + bproj[col];
        }
}

// ---------------------------------------------------------------------------
extern "C" void kernel_launch(void* const* d_in, const int* in_sizes, int n_in,
                              void* d_out, int out_size, void* d_ws, size_t ws_size,
                              hipStream_t stream)
{
    const float* x     = (const float*)d_in[0];
    const float* Wqkv  = (const float*)d_in[1];
    const float* Wproj = (const float*)d_in[2];
    const float* bproj = (const float*)d_in[3];
    const float* cosb  = (const float*)d_in[4];
    const float* sinb  = (const float*)d_in[5];
    char* ws = (char*)d_ws;
    const size_t SZ = (size_t)BB * NH * TT * HD * sizeof(_Float16);  // 8 MB
    _Float16* qf   = (_Float16*)(ws);
    _Float16* kf   = (_Float16*)(ws + SZ);
    _Float16* vf   = (_Float16*)(ws + 2 * SZ);
    _Float16* attn = (_Float16*)(ws + 3 * SZ);
    float* out = (float*)d_out;

    dim3 g1(BB * TT / 64, 3 * EMB / 64);
    qkv_gemm_kernel<<<g1, 128, 0, stream>>>(x, Wqkv, qf, kf, vf);

    rope_kernel<<<(BB * NH * TT * 32) / 256, 256, 0, stream>>>(qf, kf, cosb, sinb);

    dim3 g3(TT / 64, BB * NH);
    flash_attn_kernel<<<g3, 128, 0, stream>>>(qf, kf, vf, attn);

    dim3 g4(BB * TT / 64, EMB / 64);
    proj_gemm_kernel<<<g4, 128, 0, stream>>>(attn, Wproj, bproj, out);
}
